// TemporalGCN_50285477101897
// MI455X (gfx1250) — compile-verified, matrix operand order checked by reading, not run on hardware
//
#include <hip/hip_runtime.h>
#include <hip/hip_bf16.h>

// ---------------- problem constants (match reference) ----------------
#define NN     20000
#define EE     640000
#define D_IN   256
#define HEADS  4
#define CC     64
#define HC     256      // HEADS*CC
#define LH     64       // LSTM hidden
#define G4     256      // 4*LH
#define OUTD   128

typedef unsigned short ushort_t;
typedef __attribute__((ext_vector_type(16))) __bf16 v16bf;
typedef __attribute__((ext_vector_type(8)))  float  v8f;

union FragU { uint4 u4[2]; v16bf v; };

__device__ __forceinline__ ushort_t f2bf(float f) {
    unsigned u = __float_as_uint(f);
    unsigned r = u + 0x7FFFu + ((u >> 16) & 1u);
    return (ushort_t)(r >> 16);
}

// monotonic encoding of float for unsigned atomicMax
__device__ __forceinline__ unsigned fenc(float f) {
    unsigned u = __float_as_uint(f);
    return (u & 0x80000000u) ? ~u : (u | 0x80000000u);
}
__device__ __forceinline__ float fdec(unsigned e) {
    unsigned u = (e & 0x80000000u) ? (e & 0x7FFFFFFFu) : ~e;
    return __uint_as_float(u);
}

__device__ __forceinline__ float sigmf(float x) { return 1.0f / (1.0f + __expf(-x)); }

// ---------------- elementwise conversions / packing ----------------

__global__ void k_cvt_bf16(const float* __restrict__ src, ushort_t* __restrict__ dst, int n) {
    int i = blockIdx.x * blockDim.x + threadIdx.x;
    if (i < n) dst[i] = f2bf(src[i]);
}

// dst[j*K + k] = bf16(src[j*sj + k*sk])   (packs weight as B-matrix rows [Nout][K])
__global__ void k_pack_w(const float* __restrict__ src, ushort_t* __restrict__ dst,
                         int Nr, int K, int sj, int sk) {
    int i = blockIdx.x * blockDim.x + threadIdx.x;
    if (i >= Nr * K) return;
    int j = i / K, k = i % K;
    dst[i] = f2bf(src[(size_t)j * sj + (size_t)k * sk]);
}

// ---------------- WMMA bf16 GEMM:  C[M,Nout] = A[M,K] * B^T + bias ----------------
// A: bf16 row-major [M,K];  B: bf16 row-major [Nout,K] (pre-packed transposed weight)
// One wave computes a 16x64 output tile (4 accumulators, A fragment reused).
// act: 0 = none, 1 = relu
__global__ void k_wmma_gemm(const ushort_t* __restrict__ A, const ushort_t* __restrict__ B,
                            const float* __restrict__ bias, float* __restrict__ C,
                            int M, int K, int Nout, int act) {
    const int tilesPerRow = Nout >> 6;               // 64-wide tiles
    const int totalTiles  = (M >> 4) * tilesPerRow;
    int wid = (blockIdx.x * blockDim.x + threadIdx.x) >> 5;
    if (wid >= totalTiles) return;                   // whole-wave exit: EXEC stays full
    const int m0 = (wid / tilesPerRow) << 4;
    const int n0 = (wid % tilesPerRow) << 6;
    const int lane = threadIdx.x & 31;
    const int half = lane >> 4;                      // K-half select per ISA layout
    const int r    = lane & 15;                      // row within fragment

    v8f acc[4];
    for (int t = 0; t < 4; ++t)
        for (int i = 0; i < 8; ++i) acc[t][i] = 0.0f;

    const ushort_t* aRow = A + (size_t)(m0 + r) * K;
    for (int k0 = 0; k0 < K; k0 += 32) {
        const ushort_t* ap = aRow + k0 + half * 8;
        __builtin_prefetch((const void*)(ap + 32), 0, 0);
        FragU fa;
        fa.u4[0] = *reinterpret_cast<const uint4*>(ap);
        fa.u4[1] = *reinterpret_cast<const uint4*>(ap + 16);
#pragma unroll
        for (int t = 0; t < 4; ++t) {
            const ushort_t* bp = B + (size_t)(n0 + t * 16 + r) * K + k0 + half * 8;
            FragU fb;
            fb.u4[0] = *reinterpret_cast<const uint4*>(bp);
            fb.u4[1] = *reinterpret_cast<const uint4*>(bp + 16);
            acc[t] = __builtin_amdgcn_wmma_f32_16x16x32_bf16(
                false, fa.v, false, fb.v, (short)0, acc[t], false, false);
        }
    }
#pragma unroll
    for (int v = 0; v < 8; ++v) {
        int row = m0 + half * 8 + v;
#pragma unroll
        for (int t = 0; t < 4; ++t) {
            int col = n0 + t * 16 + r;
            float val = acc[t][v] + bias[col];
            if (act == 1) val = fmaxf(val, 0.0f);
            C[(size_t)row * Nout + col] = val;
        }
    }
}

// ---------------- edge_attr mean (single block) ----------------
__global__ void k_ea_mean(const float* __restrict__ ea, float* __restrict__ out) {
    __shared__ float red[256];
    float s = 0.0f;
    for (int i = threadIdx.x; i < EE; i += 256) s += ea[i];
    red[threadIdx.x] = s;
    __syncthreads();
    for (int w = 128; w > 0; w >>= 1) {
        if (threadIdx.x < w) red[threadIdx.x] += red[threadIdx.x + w];
        __syncthreads();
    }
    if (threadIdx.x == 0) out[0] = red[0] / (float)EE;
}

// ---------------- GATv2 attention scores + segment max ----------------
// one thread per (edge, head); edges [0,EE) real, [EE, EE+NN) self-loops
__global__ void k_score_max(const float* __restrict__ xl, const float* __restrict__ xr,
                            const float* __restrict__ ea, const float* __restrict__ We,
                            const float* __restrict__ att,
                            const int* __restrict__ src, const int* __restrict__ dst,
                            const float* __restrict__ eamean,
                            float* __restrict__ score, unsigned* __restrict__ mEnc) {
    int idx = blockIdx.x * blockDim.x + threadIdx.x;
    if (idx >= (EE + NN) * HEADS) return;
    int e = idx >> 2, h = idx & 3;
    int s, d; float a;
    if (e < EE) { s = src[e]; d = dst[e]; a = ea[e]; }
    else        { s = d = e - EE;         a = eamean[0]; }
    const float* pl = xl + (size_t)s * HC + h * CC;
    const float* pr = xr + (size_t)d * HC + h * CC;
    const float* pw = We + h * CC;
    const float* pa = att + h * CC;
    float acc = 0.0f;
#pragma unroll 8
    for (int c = 0; c < CC; ++c) {
        float z = pl[c] + pr[c] + a * pw[c];
        z = (z > 0.0f) ? z : 0.2f * z;        // leaky_relu(0.2)
        acc += pa[c] * z;
    }
    score[idx] = acc;
    atomicMax(&mEnc[d * HEADS + h], fenc(acc));
}

// exp(score - max) and segment-sum of denominators
__global__ void k_escore(const float* __restrict__ score, const unsigned* __restrict__ mEnc,
                         const int* __restrict__ dst, float* __restrict__ esc,
                         float* __restrict__ den) {
    int idx = blockIdx.x * blockDim.x + threadIdx.x;
    if (idx >= (EE + NN) * HEADS) return;
    int e = idx >> 2, h = idx & 3;
    int d = (e < EE) ? dst[e] : (e - EE);
    float m  = fdec(mEnc[d * HEADS + h]);
    float es = __expf(score[idx] - m);
    esc[idx] = es;
    atomicAdd(&den[d * HEADS + h], es);
}

// out[dst] += alpha * x_l[src]   (one block per edge, 256 channels)
__global__ void k_aggregate(const float* __restrict__ xl, const float* __restrict__ esc,
                            const float* __restrict__ den,
                            const int* __restrict__ src, const int* __restrict__ dst,
                            float* __restrict__ agg) {
    int e = blockIdx.x;
    int j = threadIdx.x;                       // 0..255
    int s, d;
    if (e < EE) { s = src[e]; d = dst[e]; } else { s = d = e - EE; }
    int h = j >> 6;
    float al = esc[e * HEADS + h] / den[d * HEADS + h];
    atomicAdd(&agg[(size_t)d * HC + j], al * xl[(size_t)s * HC + j]);
}

// elu(agg + b_gat) -> bf16 features for the LSTM input GEMMs
__global__ void k_elu_pack(const float* __restrict__ agg, const float* __restrict__ bgat,
                           ushort_t* __restrict__ hbf) {
    int i = blockIdx.x * blockDim.x + threadIdx.x;
    if (i >= NN * HC) return;
    float v = agg[i] + bgat[i & (HC - 1)];
    float hv = (v > 0.0f) ? v : (__expf(v) - 1.0f);   // elu, alpha=1
    hbf[i] = f2bf(hv);
}

// ---------------- sequential bidirectional LSTM (2 blocks, 1 per direction) ----------------
__global__ void k_lstm(const float* __restrict__ Gf, const float* __restrict__ Gb,
                       const float* __restrict__ Whh_f, const float* __restrict__ Whh_b,
                       float* __restrict__ hf, float* __restrict__ hb) {
    __shared__ float hsh[LH];
    __shared__ float gsh[G4];
    const int dir = blockIdx.x;
    const float* G = dir ? Gb : Gf;
    const float* W = dir ? Whh_b : Whh_f;
    float* out     = dir ? hb : hf;
    const int j = threadIdx.x;                 // 0..255, one gate each
    float wrow[LH];
#pragma unroll 8
    for (int k = 0; k < LH; ++k) wrow[k] = W[j * LH + k];
    if (j < LH) hsh[j] = 0.0f;
    float c = 0.0f;
    __syncthreads();
    for (int t = 0; t < NN; ++t) {
        int row = dir ? (NN - 1 - t) : t;
        float g = G[(size_t)row * G4 + j];
#pragma unroll 8
        for (int k = 0; k < LH; ++k) g += wrow[k] * hsh[k];
        gsh[j] = g;
        __syncthreads();
        float hnew = 0.0f;
        if (j < LH) {
            float ig = sigmf(gsh[j]);
            float fg = sigmf(gsh[LH + j]);
            float gg = tanhf(gsh[2 * LH + j]);
            float og = sigmf(gsh[3 * LH + j]);
            c = fg * c + ig * gg;
            hnew = og * tanhf(c);
            out[(size_t)row * LH + j] = hnew;
        }
        __syncthreads();
        if (j < LH) hsh[j] = hnew;
        __syncthreads();
    }
}

// concat [hf | hb] -> bf16 [NN, 2*LH]
__global__ void k_pack_hh(const float* __restrict__ hf, const float* __restrict__ hb,
                          ushort_t* __restrict__ hh) {
    int i = blockIdx.x * blockDim.x + threadIdx.x;
    if (i >= NN * 2 * LH) return;
    int n = i >> 7, j = i & 127;
    float v = (j < LH) ? hf[(size_t)n * LH + j] : hb[(size_t)n * LH + (j - LH)];
    hh[i] = f2bf(v);
}

// ---------------- host orchestration ----------------

static inline size_t alignup(size_t x) { return (x + 255) & ~size_t(255); }

extern "C" void kernel_launch(void* const* d_in, const int* in_sizes, int n_in,
                              void* d_out, int out_size, void* d_ws, size_t ws_size,
                              hipStream_t stream) {
    (void)in_sizes; (void)n_in; (void)out_size; (void)ws_size;

    const float* x    = (const float*)d_in[0];
    const float* ea   = (const float*)d_in[1];
    const float* Wl   = (const float*)d_in[2];
    const float* bl   = (const float*)d_in[3];
    const float* Wr   = (const float*)d_in[4];
    const float* br   = (const float*)d_in[5];
    const float* We   = (const float*)d_in[6];
    const float* att  = (const float*)d_in[7];
    const float* bgat = (const float*)d_in[8];
    const float* WihF = (const float*)d_in[9];
    const float* WhhF = (const float*)d_in[10];
    const float* bF   = (const float*)d_in[11];
    const float* WihB = (const float*)d_in[12];
    const float* WhhB = (const float*)d_in[13];
    const float* bB   = (const float*)d_in[14];
    const float* Wfc  = (const float*)d_in[15];
    const float* bfc  = (const float*)d_in[16];
    const int*   ei   = (const int*)d_in[17];
    const int* src = ei;
    const int* dst = ei + EE;
    float* outp = (float*)d_out;

    // workspace layout
    char* w = (char*)d_ws;
    size_t off = 0;
#define WS_ALLOC(name, type, bytes) type* name = (type*)(w + off); off = alignup(off + (size_t)(bytes))
    WS_ALLOC(XBF,   ushort_t, (size_t)NN * D_IN * 2);
    WS_ALLOC(WLT,   ushort_t, (size_t)HC * D_IN * 2);
    WS_ALLOC(WRT,   ushort_t, (size_t)HC * D_IN * 2);
    WS_ALLOC(WIHFT, ushort_t, (size_t)G4 * HC * 2);
    WS_ALLOC(WIHBT, ushort_t, (size_t)G4 * HC * 2);
    WS_ALLOC(WFCT,  ushort_t, (size_t)OUTD * (2 * LH) * 2);
    WS_ALLOC(XL,    float,    (size_t)NN * HC * 4);       // later reused as Gf
    WS_ALLOC(XR,    float,    (size_t)NN * HC * 4);       // later reused as Gb
    WS_ALLOC(SCORE, float,    (size_t)(EE + NN) * HEADS * 4);  // later reused as hf
    WS_ALLOC(ESC,   float,    (size_t)(EE + NN) * HEADS * 4);  // later reused as hb
    WS_ALLOC(MENC,  unsigned, (size_t)NN * HEADS * 4);
    WS_ALLOC(DEN,   float,    (size_t)NN * HEADS * 4);
    WS_ALLOC(AGG,   float,    (size_t)NN * HC * 4);       // later reused as hh bf16
    WS_ALLOC(HBF,   ushort_t, (size_t)NN * HC * 2);
    WS_ALLOC(EAM,   float,    256);
#undef WS_ALLOC

    float* Gf = XL;
    float* Gb = XR;
    float* hf = SCORE;
    float* hb = ESC;
    ushort_t* HH = (ushort_t*)AGG;

    // zero the atomic targets
    hipMemsetAsync(MENC, 0, (size_t)NN * HEADS * 4, stream);   // enc-bottom
    hipMemsetAsync(DEN,  0, (size_t)NN * HEADS * 4, stream);
    hipMemsetAsync(AGG,  0, (size_t)NN * HC * 4, stream);

    // edge-attr mean
    k_ea_mean<<<1, 256, 0, stream>>>(ea, EAM);

    // bf16 conversions / weight packs
    {
        int n = NN * D_IN;
        k_cvt_bf16<<<(n + 255) / 256, 256, 0, stream>>>(x, XBF, n);
    }
    k_pack_w<<<(HC * D_IN + 255) / 256, 256, 0, stream>>>(Wl,  WLT,  HC, D_IN, 1, HC);       // Wl[k,j] -> [j,k]
    k_pack_w<<<(HC * D_IN + 255) / 256, 256, 0, stream>>>(Wr,  WRT,  HC, D_IN, 1, HC);
    k_pack_w<<<(G4 * HC + 255) / 256, 256, 0, stream>>>(WihF, WIHFT, G4, HC, HC, 1);          // already [j,k]
    k_pack_w<<<(G4 * HC + 255) / 256, 256, 0, stream>>>(WihB, WIHBT, G4, HC, HC, 1);
    k_pack_w<<<(OUTD * 2 * LH + 255) / 256, 256, 0, stream>>>(Wfc, WFCT, OUTD, 2 * LH, 1, OUTD); // Wfc[k,j] -> [j,k]

    // GAT linear transforms (WMMA)
    {
        int tiles = (NN / 16) * (HC / 64);
        int blocks = (tiles + 7) / 8;
        k_wmma_gemm<<<blocks, 256, 0, stream>>>(XBF, WLT, bl, XL, NN, D_IN, HC, 0);
        k_wmma_gemm<<<blocks, 256, 0, stream>>>(XBF, WRT, br, XR, NN, D_IN, HC, 0);
    }

    // attention scores + segment max
    {
        int n = (EE + NN) * HEADS;
        k_score_max<<<(n + 255) / 256, 256, 0, stream>>>(XL, XR, ea, We, att, src, dst,
                                                         EAM, SCORE, MENC);
        k_escore<<<(n + 255) / 256, 256, 0, stream>>>(SCORE, MENC, dst, ESC, DEN);
    }

    // weighted aggregation (segment sum)
    k_aggregate<<<EE + NN, 256, 0, stream>>>(XL, ESC, DEN, src, dst, AGG);

    // elu + bf16 pack of GAT output
    k_elu_pack<<<(NN * HC + 255) / 256, 256, 0, stream>>>(AGG, bgat, HBF);

    // LSTM input projections hoisted out of the scan (WMMA), bias folded in
    {
        int tiles = (NN / 16) * (G4 / 64);
        int blocks = (tiles + 7) / 8;
        k_wmma_gemm<<<blocks, 256, 0, stream>>>(HBF, WIHFT, bF, Gf, NN, HC, G4, 0);
        k_wmma_gemm<<<blocks, 256, 0, stream>>>(HBF, WIHBT, bB, Gb, NN, HC, G4, 0);
    }

    // sequential recurrences, forward + backward concurrently
    k_lstm<<<2, 256, 0, stream>>>(Gf, Gb, WhhF, WhhB, hf, hb);

    // concat + final FC with relu (WMMA) straight into d_out
    k_pack_hh<<<(NN * 2 * LH + 255) / 256, 256, 0, stream>>>(hf, hb, HH);
    {
        int tiles = (NN / 16) * (OUTD / 64);
        int blocks = (tiles + 7) / 8;
        k_wmma_gemm<<<blocks, 256, 0, stream>>>(HH, WFCT, bfc, outp, NN, 2 * LH, OUTD, 1);
    }
}
